// MSRL_6305011991198
// MI455X (gfx1250) — matrix operand order repaired and checked by reading, not compiled, not run on hardware
//
#include <hip/hip_runtime.h>
#include <hip/hip_bf16.h>
#include <math.h>
#include <stdint.h>

// Problem constants (match reference)
#define NN 4096
#define PP 32768
#define EE 4096
#define DD 128
#define TT 3
#define RR 4
#define KK 16
#define WORDS (NN / 32)      // 128 u32 words per bitset row
#define CUR_TIME 1.0f

// fragment geometry: per weight matrix (128x128): 8 colTiles x 4 kChunks x 32 lanes x 16 halves
#define FRAG_HALVES_PER_MAT (8 * 4 * 32 * 16)   // 16384 halves = 32 KB
#define FRAG_BYTES_PER_MAT  (FRAG_HALVES_PER_MAT * 2)

typedef __attribute__((ext_vector_type(16))) _Float16 v16h;
typedef __attribute__((ext_vector_type(8)))  float    v8f;

// ---------------------------------------------------------------------------
// WMMA A-fragment loader (CDNA5 ISA 7.12.2, wave32, 16x16x32 f16)
// A (16x32, MxK): lanes 0-15 -> M=lane, K {0..7,16..23}; lanes 16-31 -> K {8..15,24..31}
// ---------------------------------------------------------------------------
__device__ inline v16h load_frag_a_f32(const float* A, int lda, int k0, int lane) {
    int m  = lane & 15;
    int kb = (lane >= 16) ? 8 : 0;
    v16h f;
#pragma unroll
    for (int v = 0; v < 8; ++v) {
        int base = ((v < 4) ? 2 * v : 16 + 2 * (v - 4)) + kb + k0;
        f[2 * v]     = (_Float16)A[(long)m * lda + base];
        f[2 * v + 1] = (_Float16)A[(long)m * lda + base + 1];
    }
    return f;
}

// async-stage one 32KB packed weight matrix into LDS at lds_base.
// Passing lds_base (derived from ptrtoint of the __shared__ array) captures the
// LDS object so the "memory" clobber really covers it — without this, LLVM
// proves the asm can't touch sB and folds all sB reads to undef.
__device__ inline void stage_wmat_async(const _Float16* src, unsigned lds_base, int tid) {
#pragma unroll
    for (int it = 0; it < 8; ++it) {
        unsigned goff = (unsigned)(tid * 16 + it * 4096);
        unsigned loff = lds_base + goff;
        asm volatile("global_load_async_to_lds_b128 %0, %1, %2"
                     :: "v"(loff), "v"(goff), "s"(src) : "memory");
    }
    asm volatile("s_wait_asynccnt 0x0" ::: "memory");
}

// ---------------------------------------------------------------------------
// K0: pack adjacency rows into bitsets (N x 128 u32)
// ---------------------------------------------------------------------------
__global__ void k_pack_adj(const int* __restrict__ adj, unsigned* __restrict__ bits) {
    int idx = blockIdx.x * blockDim.x + threadIdx.x;
    if (idx >= NN * WORDS) return;
    int i = idx / WORDS, w = idx % WORDS;
    const int* row = adj + (long)i * NN + w * 32;
    unsigned m = 0;
#pragma unroll 8
    for (int b = 0; b < 32; ++b)
        if (row[b] != 0) m |= (1u << b);
    bits[idx] = m;
}

// ---------------------------------------------------------------------------
// K1: sum_alpha = sum(exp(-theta * (t_now - t_events)))   (single block)
// ---------------------------------------------------------------------------
__global__ void k_sum_alpha(const float* __restrict__ t_events,
                            const float* __restrict__ theta,
                            float* __restrict__ out) {
    __shared__ float s[256];
    float th = theta[0];
    float acc = 0.0f;
    for (int i = threadIdx.x; i < EE; i += 256)
        acc += expf(-th * (CUR_TIME - t_events[i]));
    s[threadIdx.x] = acc;
    __syncthreads();
    for (int st = 128; st > 0; st >>= 1) {
        if ((int)threadIdx.x < st) s[threadIdx.x] += s[threadIdx.x + st];
        __syncthreads();
    }
    if (threadIdx.x == 0) out[0] = s[0];
}

// ---------------------------------------------------------------------------
// K2: clustering coefficient via bitset popcounts (block of 128 threads per node)
// tri[i] = sum_{j in nbr(i)} |nbr(i) & nbr(j)| ; C = 2*tri / (deg*(deg-1))
// ---------------------------------------------------------------------------
__global__ void k_tri_C(const unsigned* __restrict__ bits, float* __restrict__ Cn) {
    int i = blockIdx.x;
    int t = threadIdx.x;
    __shared__ unsigned rowi[WORDS];
    rowi[t] = bits[(long)i * WORDS + t];
    __syncthreads();
    unsigned my = rowi[t];
    int degLocal = __popc(my);
    long long tri = 0;
    for (int wb = 0; wb < WORDS; ++wb) {
        unsigned m = rowi[wb];
        while (m) {
            int b = __ffs(m) - 1;
            m &= m - 1;
            int j = wb * 32 + b;
            tri += __popc(my & bits[(long)j * WORDS + t]);
        }
    }
    __shared__ long long sT[WORDS];
    __shared__ int sD[WORDS];
    sT[t] = tri; sD[t] = degLocal;
    __syncthreads();
    for (int s = 64; s > 0; s >>= 1) {
        if (t < s) { sT[t] += sT[t + s]; sD[t] += sD[t + s]; }
        __syncthreads();
    }
    if (t == 0) {
        float deg   = (float)sD[0];
        float denom = deg * (deg - 1.0f);
        if (denom == 0.0f) denom = 1e-6f;
        Cn[i] = 2.0f * (float)sT[0] / denom;
    }
}

// ---------------------------------------------------------------------------
// K3: pack W_proj (T mats) and W_beta (R mats) fp32 -> f16 B-fragment layout.
// Record layout: frag[mat][ (colT*4+kc)*32 + lane ][16 halves]  (32B per record)
// B (32x16, KxN): lanes 0-15 -> N=lane, K {0..7,16..23} ; lanes 16-31 -> N=lane-16, K {8..15,24..31}
// ---------------------------------------------------------------------------
__global__ void k_pack_wfrag(const float* __restrict__ Wp, const float* __restrict__ Wb,
                             _Float16* __restrict__ frag) {
    int idx = blockIdx.x * blockDim.x + threadIdx.x;
    if (idx >= (TT + RR) * 8 * 4 * 32) return;
    int lane = idx & 31;
    int kc   = (idx >> 5) & 3;
    int colT = (idx >> 7) & 7;
    int m    = idx >> 10;
    const float* B = (m < TT ? Wp + (long)m * DD * DD
                             : Wb + (long)(m - TT) * DD * DD) + colT * 16;
    int n  = lane & 15;
    int kb = (lane >= 16) ? 8 : 0;
    _Float16* dst = frag + ((long)m * 1024 + (colT * 4 + kc) * 32 + lane) * 16;
#pragma unroll
    for (int v = 0; v < 8; ++v) {
        int k = kc * 32 + kb + ((v < 4) ? 2 * v : 16 + 2 * (v - 4));
        dst[2 * v]     = (_Float16)B[(long)k * DD + n];
        dst[2 * v + 1] = (_Float16)B[(long)(k + 1) * DD + n];
    }
}

// ---------------------------------------------------------------------------
// K4: proj_all[t] = node_embeds @ W_proj[t] via WMMA.
// 256 threads = 8 waves; W_proj[t] frags async-staged to LDS (32KB);
// each wave owns one 16-row tile, sweeps all 8 column tiles (32 WMMA/wave).
// ---------------------------------------------------------------------------
__global__ void k_proj_wmma(const float* __restrict__ emb,
                            const _Float16* __restrict__ wfrag,
                            float* __restrict__ proj_all) {
    __shared__ _Float16 sB[FRAG_HALVES_PER_MAT];   // 32 KB
    const int blkPerT = NN / 16 / 8;               // 32
    int t      = blockIdx.x / blkPerT;
    int rowBlk = blockIdx.x % blkPerT;
    int tid    = threadIdx.x;
    unsigned lds_base = (unsigned)(uintptr_t)(void*)&sB[0];   // low 32b of flat = LDS offset

    stage_wmat_async(wfrag + (long)t * FRAG_HALVES_PER_MAT, lds_base, tid);
    __syncthreads();

    int wave = tid >> 5, lane = tid & 31;
    int rowT = rowBlk * 8 + wave;
    const float* A = emb + (long)(rowT * 16) * DD;
    v16h a[4];
#pragma unroll
    for (int kc = 0; kc < 4; ++kc) a[kc] = load_frag_a_f32(A, DD, kc * 32, lane);

    int n  = lane & 15;
    int mb = (lane >= 16) ? 8 : 0;
    for (int colT = 0; colT < 8; ++colT) {
        v8f acc = {};
#pragma unroll
        for (int kc = 0; kc < 4; ++kc) {
            v16h b = *(const v16h*)&sB[((colT * 4 + kc) * 32 + lane) * 16];
            acc = __builtin_amdgcn_wmma_f32_16x16x32_f16(false, a[kc], false, b,
                                                         (short)0, acc, false, false);
        }
        float* Dst = proj_all + (long)t * NN * DD + (long)(rowT * 16) * DD + colT * 16;
#pragma unroll
        for (int v = 0; v < 8; ++v)
            Dst[(long)(v + mb) * DD + n] = acc[v];
    }
}

// ---------------------------------------------------------------------------
// K5: h_r[n,r,:] = mean_k node_embeds[neigh_idx[n,r,k], :]
// ---------------------------------------------------------------------------
__global__ void k_hr(const int* __restrict__ neigh, const float* __restrict__ emb,
                     float* __restrict__ h_r) {
    int nr = blockIdx.x;
    int d  = threadIdx.x;
    const int* idx = neigh + (long)nr * KK;
    float acc = 0.0f;
#pragma unroll
    for (int k = 0; k < KK; ++k)
        acc += emb[(long)idx[k] * DD + d];
    h_r[(long)nr * DD + d] = acc * (1.0f / (float)KK);
}

// ---------------------------------------------------------------------------
// K6: eps = sigmoid( mean_r( h_r[:,r,:] @ W_beta[r] + b_beta[r] ) )
// 8 waves/block; loop r: async-stage W_beta[r] frags to LDS, accumulate all 8
// column tiles into 8 persistent accumulators (128 WMMA per wave total).
// ---------------------------------------------------------------------------
__global__ void k_eps_wmma(const float* __restrict__ h_r,
                           const _Float16* __restrict__ wfrag_beta,
                           const float* __restrict__ b_beta,
                           float* __restrict__ eps) {
    __shared__ _Float16 sB[FRAG_HALVES_PER_MAT];   // 32 KB
    int rowBlk = blockIdx.x;
    int tid    = threadIdx.x;
    int wave   = tid >> 5, lane = tid & 31;
    int rowT   = rowBlk * 8 + wave;
    unsigned lds_base = (unsigned)(uintptr_t)(void*)&sB[0];

    v8f acc[8];
#pragma unroll
    for (int c = 0; c < 8; ++c) acc[c] = (v8f){};

    for (int r = 0; r < RR; ++r) {
        __syncthreads();   // previous-iteration LDS reads complete
        stage_wmat_async(wfrag_beta + (long)r * FRAG_HALVES_PER_MAT, lds_base, tid);
        __syncthreads();

        // A row m = h_r[(rowT*16+m)*R*D + r*D + k]  -> lda = R*D
        const float* A = h_r + ((long)(rowT * 16) * RR + r) * DD;
        v16h a[4];
#pragma unroll
        for (int kc = 0; kc < 4; ++kc) a[kc] = load_frag_a_f32(A, RR * DD, kc * 32, lane);

#pragma unroll
        for (int colT = 0; colT < 8; ++colT) {
#pragma unroll
            for (int kc = 0; kc < 4; ++kc) {
                v16h b = *(const v16h*)&sB[((colT * 4 + kc) * 32 + lane) * 16];
                acc[colT] = __builtin_amdgcn_wmma_f32_16x16x32_f16(false, a[kc], false, b,
                                                                   (short)0, acc[colT],
                                                                   false, false);
            }
        }
    }

    int n  = lane & 15;
    int mb = (lane >= 16) ? 8 : 0;
#pragma unroll
    for (int colT = 0; colT < 8; ++colT) {
        float bias = 0.0f;
#pragma unroll
        for (int r = 0; r < RR; ++r) bias += b_beta[r * DD + colT * 16 + n];
#pragma unroll
        for (int v = 0; v < 8; ++v) {
            float x = (acc[colT][v] + bias) * (1.0f / (float)RR);
            eps[(long)(rowT * 16 + v + mb) * DD + colT * 16 + n] =
                1.0f / (1.0f + expf(-x));
        }
    }
}

// ---------------------------------------------------------------------------
// K7: pair stage — one wave32 per pair
// ---------------------------------------------------------------------------
__global__ void k_pairs(const int* __restrict__ pairs, const int* __restrict__ node_type,
                        const float* __restrict__ proj_all, const float* __restrict__ emb,
                        const unsigned* __restrict__ bits, const float* __restrict__ Cn,
                        const float* __restrict__ eps, const float* __restrict__ s_alpha,
                        const float* __restrict__ q1p, const float* __restrict__ q2p,
                        float* __restrict__ out) {
    int gtid = blockIdx.x * blockDim.x + threadIdx.x;
    int wid  = gtid >> 5;
    int lane = threadIdx.x & 31;
    if (wid >= PP) return;

    int pm = pairs[2 * wid], pn = pairs[2 * wid + 1];
    int tm = node_type[pm], tn = node_type[pn];
    const float* projm = proj_all + ((long)tm * NN + pm) * DD;
    const float* projn = proj_all + ((long)tn * NN + pn) * DD;
    const float* em  = emb + (long)pm * DD;
    const float* en  = emb + (long)pn * DD;
    const float* epm = eps + (long)pm * DD;
    const float* epn = eps + (long)pn * DD;

    float dp = 0.0f, de = 0.0f, dn = 0.0f;
    for (int d = lane; d < DD; d += 32) {
        float a = projm[d] - projn[d]; dp += a * a;
        float b = em[d] - en[d];       de += b * b;
        dn += epm[d] * epn[d];
    }

    float cs = 0.0f;
    const unsigned* bm = bits + (long)pm * WORDS;
    const unsigned* bn = bits + (long)pn * WORDS;
    for (int w = lane; w < WORDS; w += 32) {
        unsigned m = bm[w] & bn[w];
        while (m) {
            int b = __ffs(m) - 1;
            m &= m - 1;
            cs += Cn[w * 32 + b];
        }
    }

    for (int off = 16; off > 0; off >>= 1) {
        dp += __shfl_down(dp, off);
        de += __shfl_down(de, off);
        dn += __shfl_down(dn, off);
        cs += __shfl_down(cs, off);
    }

    if (lane == 0) {
        float l_hist  = -sqrtf(dp) + s_alpha[0];
        float l_tri   = cs * (-de);
        float l_neigh = dn * (1.0f / (float)DD);
        float lam = expf(l_hist + l_tri + l_neigh);
        float z   = q1p[0] * lam + q2p[0];
        out[wid]  = 1.0f / (1.0f + expf(-z));
    }
}

// ---------------------------------------------------------------------------
extern "C" void kernel_launch(void* const* d_in, const int* in_sizes, int n_in,
                              void* d_out, int out_size, void* d_ws, size_t ws_size,
                              hipStream_t stream) {
    const int*   pairs     = (const int*)d_in[0];
    const int*   adj       = (const int*)d_in[1];
    const int*   neigh     = (const int*)d_in[2];
    const int*   node_type = (const int*)d_in[3];
    const float* t_events  = (const float*)d_in[4];
    const float* emb       = (const float*)d_in[5];
    const float* W_proj    = (const float*)d_in[6];
    const float* W_beta    = (const float*)d_in[7];
    const float* b_beta    = (const float*)d_in[8];
    const float* theta     = (const float*)d_in[9];
    const float* q1        = (const float*)d_in[10];
    const float* q2        = (const float*)d_in[11];
    float* out = (float*)d_out;

    // workspace carve-up (~18.2 MB total)
    char* ws = (char*)d_ws;
    unsigned* bits     = (unsigned*)ws;  ws += (size_t)NN * WORDS * 4;              // 2 MB
    float*    Cn       = (float*)ws;     ws += (size_t)NN * 4;                      // 16 KB
    float*    proj_all = (float*)ws;     ws += (size_t)TT * NN * DD * 4;            // 6 MB
    float*    h_r      = (float*)ws;     ws += (size_t)NN * RR * DD * 4;            // 8 MB
    float*    eps      = (float*)ws;     ws += (size_t)NN * DD * 4;                 // 2 MB
    _Float16* wfrag    = (_Float16*)ws;  ws += (size_t)(TT + RR) * FRAG_BYTES_PER_MAT; // 224 KB
    float*    s_alpha  = (float*)ws;     ws += 4;

    k_pack_adj<<<(NN * WORDS + 255) / 256, 256, 0, stream>>>(adj, bits);
    k_sum_alpha<<<1, 256, 0, stream>>>(t_events, theta, s_alpha);
    k_pack_wfrag<<<((TT + RR) * 1024 + 255) / 256, 256, 0, stream>>>(W_proj, W_beta, wfrag);
    k_tri_C<<<NN, WORDS, 0, stream>>>(bits, Cn);
    k_proj_wmma<<<TT * (NN / 16 / 8), 256, 0, stream>>>(emb, wfrag, proj_all);
    k_hr<<<NN * RR, DD, 0, stream>>>(neigh, emb, h_r);
    k_eps_wmma<<<NN / 16 / 8, 256, 0, stream>>>(h_r, wfrag + (size_t)TT * FRAG_HALVES_PER_MAT,
                                                b_beta, eps);
    k_pairs<<<(PP * 32) / 256, 256, 0, stream>>>(pairs, node_type, proj_all, emb,
                                                 bits, Cn, eps, s_alpha, q1, q2, out);
}
